// ConvCNP_17952963297848
// MI455X (gfx1250) — compile-verified, matrix operand order checked by reading, not run on hardware
//
#include <hip/hip_runtime.h>
#include <hip/hip_bf16.h>

typedef __attribute__((ext_vector_type(16))) _Float16 v16h;
typedef __attribute__((ext_vector_type(8)))  float    v8f;
typedef __attribute__((ext_vector_type(4)))  float    v4f;

#define BATCH 32
#define NC    2048
#define NT    1024

// ---------------------------------------------------------------------------
// 1) global min/max over xc and xt (single workgroup, LDS tree reduce)
// ---------------------------------------------------------------------------
__global__ __launch_bounds__(1024)
void minmax_kernel(const float* __restrict__ xc, int nxc,
                   const float* __restrict__ xt, int nxt,
                   float* __restrict__ mm)
{
    __shared__ float smin[1024];
    __shared__ float smax[1024];
    int tid = threadIdx.x;
    float lo =  INFINITY, hi = -INFINITY;
    for (int i = tid; i < nxc; i += 1024) { float v = xc[i]; lo = fminf(lo, v); hi = fmaxf(hi, v); }
    for (int i = tid; i < nxt; i += 1024) { float v = xt[i]; lo = fminf(lo, v); hi = fmaxf(hi, v); }
    smin[tid] = lo; smax[tid] = hi;
    __syncthreads();
    for (int s = 512; s > 0; s >>= 1) {
        if (tid < s) {
            smin[tid] = fminf(smin[tid], smin[tid + s]);
            smax[tid] = fmaxf(smax[tid], smax[tid + s]);
        }
        __syncthreads();
    }
    if (tid == 0) { mm[0] = smin[0]; mm[1] = smax[0]; }
}

// ---------------------------------------------------------------------------
// 2) psi stage: h0 = sum_c K(t, xc), h1 = (sum_c K*yc)/(h0+1e-8)
//    writes conv input h in (B, 3, NT) layout: ch0=t, ch1=h0, ch2=h1
//    one thread per (b, t); each block covers one b (NT/256 = 4 blocks per b)
// ---------------------------------------------------------------------------
__global__ __launch_bounds__(256)
void psi_kernel(const float* __restrict__ xc, const float* __restrict__ yc,
                const float* __restrict__ ls, const float* __restrict__ os,
                const float* __restrict__ mm, float* __restrict__ h)
{
    int gid = blockIdx.x * 256 + threadIdx.x;
    int b  = gid >> 10;
    int ti = gid & (NT - 1);
    float lower = mm[0], upper = mm[1];
    float t = lower + (float)ti * ((upper - lower) * (1.0f / (NT - 1)));
    float inv_ls = 1.0f / ls[0];
    float osv = os[0];

    __shared__ float sx[256];
    __shared__ float sy[256];
    const float* xb = xc + b * NC;
    const float* yb = yc + b * NC;

    float s0 = 0.0f, s1 = 0.0f;
    for (int c0 = 0; c0 < NC; c0 += 256) {
        __syncthreads();
        sx[threadIdx.x] = xb[c0 + threadIdx.x];
        sy[threadIdx.x] = yb[c0 + threadIdx.x];
        __syncthreads();
        if (c0 + 256 < NC) {
            // hide next chunk's latency behind the exp loop (global_prefetch_b8)
            __builtin_prefetch(xb + c0 + 256 + threadIdx.x, 0, 0);
            __builtin_prefetch(yb + c0 + 256 + threadIdx.x, 0, 0);
        }
        #pragma unroll 8
        for (int c = 0; c < 256; ++c) {
            float d = (t - sx[c]) * inv_ls;
            float k = osv * __expf(-0.5f * d * d);
            s0 += k;
            s1 += k * sy[c];
        }
    }
    float h0 = s0;
    float h1 = s1 / (h0 + 1e-8f);
    float* hb = h + b * 3 * NT;
    hb[ti]          = t;
    hb[NT + ti]     = h0;
    hb[2 * NT + ti] = h1;
}

// ---------------------------------------------------------------------------
// 3) conv1d (kernel 5, pad 2) as WMMA GEMM: out[m][i] = b[m] + A(m,K) x B(K,i)
//    A = weights (COUT x CIN*5, zero-padded to KPAD), B = im2col of input.
//    Workgroup = 128 threads (4 waves); block handles (batch b, 64 columns);
//    each wave owns a 16-column N-tile. f16 inputs, f32 accumulate.
//    A fragments loaded as 2x ds_load_b128 (contiguous 8-half runs per lane).
// ---------------------------------------------------------------------------
template <int CIN, int COUT, int KDIM, int KPAD, bool RELU>
__global__ __launch_bounds__(128)
void conv_wmma(const float* __restrict__ in, const float* __restrict__ W,
               const float* __restrict__ bias, float* __restrict__ out)
{
    constexpr int MT = (COUT + 15) / 16;                    // M tiles
    __shared__ _Float16 s_in[CIN][68];                      // halo'd input tile
    __shared__ __align__(16) _Float16 s_w[MT * 16][KPAD];   // zero-padded weights

    int b  = blockIdx.x >> 4;                  // 16 tiles of 64 columns per batch
    int n0 = (blockIdx.x & 15) * 64;
    int tid = threadIdx.x;

    // stage input tile (with zero padding at the sequence edges)
    for (int idx = tid; idx < CIN * 68; idx += 128) {
        int c = idx / 68, p = idx % 68;
        int pos = n0 - 2 + p;
        float v = (pos >= 0 && pos < NT) ? in[(b * CIN + c) * NT + pos] : 0.0f;
        s_in[c][p] = (_Float16)v;
    }
    // stage weights, zero-padded in both M and K
    for (int idx = tid; idx < MT * 16 * KPAD; idx += 128) {
        int m = idx / KPAD, K = idx % KPAD;
        float v = (m < COUT && K < KDIM) ? W[m * KDIM + K] : 0.0f;
        s_w[m][K] = (_Float16)v;
    }
    __syncthreads();

    int lane  = tid & 31;
    int wave  = tid >> 5;
    int lrow  = lane & 15;                 // M-row (A) / N-col (B,C) within tile
    int khalf = (lane >> 4) * 8;           // half-wave K offset
    int nloc  = wave * 16 + lrow;          // column within the 64-wide block tile

    #pragma unroll
    for (int mt = 0; mt < MT; ++mt) {
        v8f acc = {};
        int arow = mt * 16 + lrow;
        const uint4* wrow = (const uint4*)&s_w[arow][0];   // 8 halves per uint4
        #pragma unroll
        for (int kk = 0; kk < KPAD; kk += 32) {
            // A fragment: two contiguous 16B runs -> two ds_load_b128
            union { uint4 u[2]; v16h v; } a;
            a.u[0] = wrow[(kk + khalf) >> 3];
            a.u[1] = wrow[(kk + 16 + khalf) >> 3];

            // B fragment: im2col gather (K -> (c,k); crosses channels every 5)
            v16h bv;
            #pragma unroll
            for (int e = 0; e < 16; ++e) {
                int K = kk + e + ((e >= 8) ? 8 : 0) + khalf;   // per-lane K mapping
                _Float16 bval = (_Float16)0.0f;
                if (K < KDIM) {
                    int c = K / 5, k = K % 5;
                    bval = s_in[c][nloc + k];
                }
                bv[e] = bval;
            }
            acc = __builtin_amdgcn_wmma_f32_16x16x32_f16(
                /*neg_a=*/false, a.v, /*neg_b=*/false, bv,
                /*c_mod=*/(short)0, acc, /*reuse_a=*/false, /*reuse_b=*/false);
        }
        // C layout: VGPR r -> M = r + 8*(lane>=16), N = lane%16
        int i = n0 + nloc;
        #pragma unroll
        for (int r = 0; r < 8; ++r) {
            int m = mt * 16 + r + ((lane >> 4) * 8);
            if (m < COUT) {
                float v = acc[r] + bias[m];
                if (RELU) v = fmaxf(v, 0.0f);
                out[(b * COUT + m) * NT + i] = v;
            }
        }
    }
}

// ---------------------------------------------------------------------------
// 4) rho stage: mu = sum_j K(xt, t_j) f_mu[j]; sigma = sum_j K * softplus(f_sig[j])
// ---------------------------------------------------------------------------
__device__ __forceinline__ float softplus_f(float v) {
    return (v > 20.0f) ? v : log1pf(__expf(v));
}

__global__ __launch_bounds__(256)
void rho_kernel(const float* __restrict__ xt, const float* __restrict__ z4,
                const float* __restrict__ ls, const float* __restrict__ os,
                const float* __restrict__ mm,
                float* __restrict__ mu_out, float* __restrict__ sig_out)
{
    int gid = blockIdx.x * 256 + threadIdx.x;
    int b = gid >> 10;

    __shared__ float sf[2 * NT];   // [0..NT): f_mu, [NT..2NT): softplus(f_sigma)
    for (int idx = threadIdx.x; idx < NT; idx += 256) {
        sf[idx]      = z4[(b * 2 + 0) * NT + idx];
        sf[NT + idx] = softplus_f(z4[(b * 2 + 1) * NT + idx]);
    }
    __syncthreads();

    float x = xt[gid];
    float lower = mm[0];
    float step  = (mm[1] - mm[0]) * (1.0f / (NT - 1));
    float inv_ls = 1.0f / ls[0];
    float osv = os[0];

    float mu = 0.0f, sg = 0.0f;
    #pragma unroll 4
    for (int j = 0; j < NT; ++j) {
        float tj = lower + (float)j * step;
        float d  = (x - tj) * inv_ls;
        float k  = osv * __expf(-0.5f * d * d);
        mu += k * sf[j];
        sg += k * sf[NT + j];
    }
    mu_out[gid]  = mu;
    sig_out[gid] = sg;
}

// ---------------------------------------------------------------------------
// 5) Sigma = diag(sigma): stream 134 MB of mostly zeros; non-temporal b128
//    stores (native clang vector type for the builtin) so the zero stream
//    doesn't pollute L2.
// ---------------------------------------------------------------------------
__global__ __launch_bounds__(256)
void sigma_fill(const float* __restrict__ sig, float* __restrict__ out)
{
    long long idx = (long long)blockIdx.x * 256 + threadIdx.x;  // v4f index
    long long row = idx >> 8;          // b*NT + i   (256 v4f per row)
    int i  = (int)(row & (NT - 1));
    int q  = (int)(idx & 255);         // j/4
    v4f v = {0.0f, 0.0f, 0.0f, 0.0f};
    if ((i >> 2) == q) {
        v[i & 3] = sig[row];
    }
    __builtin_nontemporal_store(v, (v4f*)out + idx);
}

// ---------------------------------------------------------------------------
// launcher
// ---------------------------------------------------------------------------
extern "C" void kernel_launch(void* const* d_in, const int* in_sizes, int n_in,
                              void* d_out, int out_size, void* d_ws, size_t ws_size,
                              hipStream_t stream) {
    const float* xc     = (const float*)d_in[0];
    const float* yc     = (const float*)d_in[1];
    const float* xt     = (const float*)d_in[2];
    const float* ls_psi = (const float*)d_in[3];
    const float* os_psi = (const float*)d_in[4];
    const float* ls_rho = (const float*)d_in[5];
    const float* os_rho = (const float*)d_in[6];
    const float* W1 = (const float*)d_in[7];
    const float* b1 = (const float*)d_in[8];
    const float* W2 = (const float*)d_in[9];
    const float* b2 = (const float*)d_in[10];
    const float* W3 = (const float*)d_in[11];
    const float* b3 = (const float*)d_in[12];
    const float* W4 = (const float*)d_in[13];
    const float* b4 = (const float*)d_in[14];

    float* out = (float*)d_out;
    float* ws  = (float*)d_ws;

    // workspace layout (floats) — total ~9.2 MB
    float* mm  = ws;                         // 2
    float* h   = ws + 16;                    // B*3*NT   = 98304
    float* z1  = h  + BATCH * 3  * NT;       // B*16*NT  = 524288
    float* z2  = z1 + BATCH * 16 * NT;       // B*32*NT  = 1048576
    float* z3  = z2 + BATCH * 32 * NT;       // B*16*NT  = 524288
    float* z4  = z3 + BATCH * 16 * NT;       // B*2*NT   = 65536
    float* sg  = z4 + BATCH * 2  * NT;       // B*NT     = 32768

    int nxc = in_sizes[0];   // 65536
    int nxt = in_sizes[2];   // 32768

    minmax_kernel<<<1, 1024, 0, stream>>>(xc, nxc, xt, nxt, mm);

    psi_kernel<<<(BATCH * NT) / 256, 256, 0, stream>>>(xc, yc, ls_psi, os_psi, mm, h);

    // conv stack: grid = B * (NT/64) blocks, 128 threads (4 waves) each
    conv_wmma< 3, 16,  15,  32, true ><<<BATCH * 16, 128, 0, stream>>>(h,  W1, b1, z1);
    conv_wmma<16, 32,  80,  96, true ><<<BATCH * 16, 128, 0, stream>>>(z1, W2, b2, z2);
    conv_wmma<32, 16, 160, 160, true ><<<BATCH * 16, 128, 0, stream>>>(z2, W3, b3, z3);
    conv_wmma<16,  2,  80,  96, false><<<BATCH * 16, 128, 0, stream>>>(z3, W4, b4, z4);

    // mu goes directly into d_out[0 .. B*NT)
    rho_kernel<<<(BATCH * NT) / 256, 256, 0, stream>>>(xt, z4, ls_rho, os_rho, mm, out, sg);

    // Sigma: B*NT*NT floats after mu
    long long nf4 = (long long)BATCH * NT * NT / 4;   // 8388608 vec4 stores
    sigma_fill<<<(unsigned)(nf4 / 256), 256, 0, stream>>>(sg, out + BATCH * NT);
}
